// TemGNN_90744069030653
// MI455X (gfx1250) — compile-verified, matrix operand order checked by reading, not run on hardware
//
#include <hip/hip_runtime.h>
#include <hip/hip_bf16.h>

#define NN_NODES 50000
#define TT_STEPS 8
#define EE_EDGES 800000
#define VOCAB    1000
#define EDIM     16
#define DDIM     32
#define G0DIM    64
#define HDIM     128

typedef __attribute__((ext_vector_type(16))) _Float16 v16h;
typedef __attribute__((ext_vector_type(8)))  _Float16 v8h;
typedef __attribute__((ext_vector_type(8)))  float    v8f;

// ---------------------------------------------------------------------------
// Convert f32 weight W[K][NN] (row-major) into WMMA B-fragment layout:
// fragment index = ((nt*(K/32)+kt)*32 + lane)*16 + i  holds B[k][n] with
//   n = nt*16 + (lane&15), kb = kt*32 + (lane<16?0:8),
//   k = kb+i (i<8) or kb+16+(i-8) (i>=8)        [CDNA5 ISA 7.12.2 16-bit B]
// blockIdx.y = batch (for per-timestep weights).
// ---------------------------------------------------------------------------
__global__ void convert_wfrag(const float* __restrict__ W, _Float16* __restrict__ out,
                              int K, int NN) {
    const size_t bo = (size_t)blockIdx.y * K * NN;
    W += bo; out += bo;
    int tid = blockIdx.x * blockDim.x + threadIdx.x;
    int ktiles = K >> 5;
    int total = (NN >> 4) * ktiles * 32;
    if (tid >= total) return;
    int lane = tid & 31;
    int kt = (tid >> 5) % ktiles;
    int nt = (tid >> 5) / ktiles;
    int n  = (nt << 4) + (lane & 15);
    int kb = (kt << 5) + ((lane < 16) ? 0 : 8);
    _Float16* o = out + (size_t)tid * 16;
#pragma unroll
    for (int i = 0; i < 8; ++i) o[i]     = (_Float16)W[(size_t)(kb + i) * NN + n];
#pragma unroll
    for (int i = 0; i < 8; ++i) o[8 + i] = (_Float16)W[(size_t)(kb + 16 + i) * NN + n];
}

// ---------------------------------------------------------------------------
// Gather-concat: out[i] = [emb0[idx0], emb1[idx1], dense[i]]  ([B,64])
// Works for static (B=N) and dynamic (B=T*N) since layouts flatten identically.
// ---------------------------------------------------------------------------
__global__ void build_concat_x(const float* __restrict__ dense,  // [B,32]
                               const int*   __restrict__ sparse, // [B,2]
                               const float* __restrict__ emb,    // [2,V,16]
                               float* __restrict__ out, int B) { // [B,64]
    int i = blockIdx.x * blockDim.x + threadIdx.x;
    if (i >= B) return;
    int i0 = sparse[2 * i], i1 = sparse[2 * i + 1];
    const float4* e0 = (const float4*)(emb + (size_t)i0 * EDIM);
    const float4* e1 = (const float4*)(emb + (size_t)VOCAB * EDIM + (size_t)i1 * EDIM);
    const float4* dn = (const float4*)(dense + (size_t)i * DDIM);
    float4* o = (float4*)(out + (size_t)i * G0DIM);
#pragma unroll
    for (int j = 0; j < 4; ++j) o[j] = e0[j];
#pragma unroll
    for (int j = 0; j < 4; ++j) o[4 + j] = e1[j];
#pragma unroll
    for (int j = 0; j < 8; ++j) o[8 + j] = dn[j];
}

// ---------------------------------------------------------------------------
// Weighted edge scatter: out[dst] += w * x[src], D floats per node.
// One thread per (edge, 16-float chunk); f32 atomics (L2-resident dest).
// ---------------------------------------------------------------------------
__global__ void scatter_edges(const float* __restrict__ x, const int* __restrict__ eidx,
                              const float* __restrict__ w, float* __restrict__ out,
                              int D, int chunkShift) {
    long long tid = (long long)blockIdx.x * blockDim.x + threadIdx.x;
    long long total = (long long)EE_EDGES << chunkShift;
    if (tid >= total) return;
    int e = (int)(tid >> chunkShift);
    int c = (int)(tid & ((1 << chunkShift) - 1));
    int s = eidx[e];
    int d = eidx[EE_EDGES + e];
    float wt = w[e];
    const float4* xs = (const float4*)(x + (size_t)s * D + c * 16);
    float* od = out + (size_t)d * D + c * 16;
#pragma unroll
    for (int j = 0; j < 4; ++j) {
        float4 v = xs[j];
        atomicAdd(od + 4 * j + 0, wt * v.x);
        atomicAdd(od + 4 * j + 1, wt * v.y);
        atomicAdd(od + 4 * j + 2, wt * v.z);
        atomicAdd(od + 4 * j + 3, wt * v.w);
    }
}

// ---------------------------------------------------------------------------
// WMMA fragment helpers (CDNA5 16-bit A layout, wave32).
// ---------------------------------------------------------------------------
__device__ __forceinline__ v16h load_a_frag(const _Float16* rowA, int kt, int lane) {
    int kb = kt * 32 + ((lane < 16) ? 0 : 8);
    v8h lo = *(const v8h*)(rowA + kb);
    v8h hi = *(const v8h*)(rowA + kb + 16);
    return __builtin_shufflevector(lo, hi, 0, 1, 2, 3, 4, 5, 6, 7,
                                            8, 9, 10, 11, 12, 13, 14, 15);
}

template <int K>
__device__ __forceinline__ void wmma_accum(const _Float16* rowA, const _Float16* Bf,
                                           int nt, int lane, v8f& acc) {
#pragma unroll
    for (int kt = 0; kt < K / 32; ++kt) {
        v16h a = load_a_frag(rowA, kt, lane);
        v16h b = *(const v16h*)(Bf + ((size_t)(nt * (K / 32) + kt) * 32 + lane) * 16);
        acc = __builtin_amdgcn_wmma_f32_16x16x32_f16(false, a, false, b,
                                                     (short)0, acc, false, false);
    }
}

// Store one 16x16 C tile (v8f layout: lane -> col lane&15, vgpr v -> row
// (lane>>4)*8 + v). Fast path: whole strip in range -> 8 unguarded stores with
// immediate row offsets (no exec-mask churn).
template <int LDC, bool RELU>
__device__ __forceinline__ void store_ctile(float* __restrict__ C, int r0, int col,
                                            int lane, const v8f& acc, bool full, int M) {
    int rbase = r0 + (lane >> 4) * 8;
    float* cp = C + (size_t)rbase * LDC + col + (lane & 15);
    if (full) {
#pragma unroll
        for (int v = 0; v < 8; ++v) {
            float o = acc[v];
            if (RELU) o = o > 0.f ? o : 0.f;
            cp[(size_t)v * LDC] = o;
        }
    } else {
#pragma unroll
        for (int v = 0; v < 8; ++v) {
            if (rbase + v < M) {
                float o = acc[v];
                if (RELU) o = o > 0.f ? o : 0.f;
                cp[(size_t)v * LDC] = o;
            }
        }
    }
}

// ---------------------------------------------------------------------------
// Generic skinny GEMM: C[M,NN] = act(A_f32[M,K] @ Bfrag_f16 + bias).
// 256 threads = 8 waves; wave owns 16 rows (block = 128 rows); A staged in LDS
// as f16 per-wave; only intra-wave LDS reuse -> s_wait_dscnt, no barrier.
// ---------------------------------------------------------------------------
template <int K, int NN, bool RELU>
__global__ __launch_bounds__(256) void gemm_bias_act(const float* __restrict__ A,
                                                     const _Float16* __restrict__ Bf,
                                                     const float* __restrict__ bias,
                                                     float* __restrict__ C, int M) {
    __shared__ _Float16 smem[8 * 16 * K];
    int wave = threadIdx.x >> 5;
    int lane = threadIdx.x & 31;
    int r0 = blockIdx.x * 128 + wave * 16;
    _Float16* tile = smem + wave * 16 * K;
    for (int idx = lane; idx < 16 * K; idx += 32) {
        int row = idx / K, col = idx % K;
        int gr = r0 + row;
        float v = (gr < M) ? A[(size_t)gr * K + col] : 0.f;
        tile[idx] = (_Float16)v;
    }
    asm volatile("s_wait_dscnt 0" ::: "memory");  // wave-local LDS RAW fence

    const _Float16* rowA = tile + (size_t)(lane & 15) * K;
    const bool full = (r0 + 16) <= M;
#pragma unroll
    for (int nt = 0; nt < NN / 16; ++nt) {
        float bv = bias[nt * 16 + (lane & 15)];
        v8f acc;
#pragma unroll
        for (int v = 0; v < 8; ++v) acc[v] = bv;
        wmma_accum<K>(rowA, Bf, nt, lane, acc);
        store_ctile<NN, RELU>(C, r0, nt * 16, lane, acc, full, M);
    }
}

// ---------------------------------------------------------------------------
// GRU fused GEMM: out[N,512] = [ xr+hr | xz+hz | xn | hn ]
//   gx = G @ Wx + bx (384 cols), gh = Hst @ Wh + bh (384 cols)
// r/z chunks share one accumulator (sum of both products); n chunk kept split.
// ---------------------------------------------------------------------------
__global__ __launch_bounds__(256) void gru_gemm(const float* __restrict__ G,
                                                const float* __restrict__ Hst,
                                                const _Float16* __restrict__ WxF,
                                                const _Float16* __restrict__ WhF,
                                                const float* __restrict__ bx,
                                                const float* __restrict__ bh,
                                                float* __restrict__ out, int M) {
    __shared__ _Float16 smem[8 * 2 * 16 * HDIM];  // 64 KB
    int wave = threadIdx.x >> 5;
    int lane = threadIdx.x & 31;
    int r0 = blockIdx.x * 128 + wave * 16;
    _Float16* tg = smem + wave * 2 * 16 * HDIM;
    _Float16* th = tg + 16 * HDIM;
    for (int idx = lane; idx < 2 * 16 * HDIM; idx += 32) {
        int row = idx / HDIM, col = idx % HDIM;
        float v;
        if (row < 16) {
            int gr = r0 + row;
            v = (gr < M) ? G[(size_t)gr * HDIM + col] : 0.f;
        } else {
            int gr = r0 + (row - 16);
            v = (gr < M) ? Hst[(size_t)gr * HDIM + col] : 0.f;
        }
        tg[idx] = (_Float16)v;
    }
    asm volatile("s_wait_dscnt 0" ::: "memory");

    const _Float16* rowG = tg + (size_t)(lane & 15) * HDIM;
    const _Float16* rowH = th + (size_t)(lane & 15) * HDIM;
    int nlo = lane & 15;
    const bool full = (r0 + 16) <= M;

    // r and z gate pre-activations (cols 0..255): gx + gh summed
#pragma unroll
    for (int nt = 0; nt < 16; ++nt) {
        float bv = bx[nt * 16 + nlo] + bh[nt * 16 + nlo];
        v8f acc;
#pragma unroll
        for (int v = 0; v < 8; ++v) acc[v] = bv;
        wmma_accum<HDIM>(rowG, WxF, nt, lane, acc);
        wmma_accum<HDIM>(rowH, WhF, nt, lane, acc);
        store_ctile<512, false>(out, r0, nt * 16, lane, acc, full, M);
    }
    // xn (cols 256..383) and hn (cols 384..511) kept separate
#pragma unroll
    for (int nt = 16; nt < 24; ++nt) {
        float bvx = bx[nt * 16 + nlo];
        float bvh = bh[nt * 16 + nlo];
        v8f ax, ah;
#pragma unroll
        for (int v = 0; v < 8; ++v) { ax[v] = bvx; ah[v] = bvh; }
        wmma_accum<HDIM>(rowG, WxF, nt, lane, ax);
        wmma_accum<HDIM>(rowH, WhF, nt, lane, ah);
        store_ctile<512, false>(out, r0, nt * 16, lane, ax, full, M);
        store_ctile<512, false>(out, r0, nt * 16 + 128, lane, ah, full, M);
    }
}

// ---------------------------------------------------------------------------
// GRU gates elementwise: h_new = (1-z)*tanh(xn + r*hn) + z*h
// ---------------------------------------------------------------------------
__global__ void gru_gates(const float* __restrict__ gxh, float* __restrict__ h,
                          float* __restrict__ hs_t, int M) {
    int i = blockIdx.x * blockDim.x + threadIdx.x;
    if (i >= M * HDIM) return;
    int n = i >> 7, d = i & 127;
    const float* g = gxh + (size_t)n * 512;
    float r  = 1.f / (1.f + __expf(-g[d]));
    float z  = 1.f / (1.f + __expf(-g[128 + d]));
    float nc = tanhf(g[256 + d] + r * g[384 + d]);
    float hn = (1.f - z) * nc + z * h[i];
    h[i] = hn;
    hs_t[i] = hn;
}

// ---------------------------------------------------------------------------
// Temporal attention: one wave per node; 9 embeds x 128 dims.
// ---------------------------------------------------------------------------
__global__ __launch_bounds__(256) void attention_out(const float* __restrict__ se,
                                                     const float* __restrict__ hs,
                                                     const float* __restrict__ Waw,
                                                     const float* __restrict__ baw,
                                                     float* __restrict__ out) {
    int wave = threadIdx.x >> 5, lane = threadIdx.x & 31;
    int n = blockIdx.x * 8 + wave;
    if (n >= NN_NODES) return;
    float wv[4], e[9][4], sc[9];
#pragma unroll
    for (int d = 0; d < 4; ++d) wv[d] = Waw[lane * 4 + d];
#pragma unroll
    for (int d = 0; d < 4; ++d) e[0][d] = se[(size_t)n * HDIM + lane * 4 + d];
#pragma unroll
    for (int t = 1; t <= TT_STEPS; ++t)
#pragma unroll
        for (int d = 0; d < 4; ++d)
            e[t][d] = hs[((size_t)(t - 1) * NN_NODES + n) * HDIM + lane * 4 + d];
#pragma unroll
    for (int t = 0; t < 9; ++t) {
        float p = 0.f;
#pragma unroll
        for (int d = 0; d < 4; ++d) p += tanhf(e[t][d]) * wv[d];
#pragma unroll
        for (int off = 16; off; off >>= 1) p += __shfl_xor(p, off, 32);
        sc[t] = p + baw[0];
    }
    float mx = sc[0];
#pragma unroll
    for (int t = 1; t < 9; ++t) mx = fmaxf(mx, sc[t]);
    float den = 0.f;
#pragma unroll
    for (int t = 0; t < 9; ++t) { sc[t] = __expf(sc[t] - mx); den += sc[t]; }
    float inv = 1.f / den;
#pragma unroll
    for (int d = 0; d < 4; ++d) {
        float o = 0.f;
#pragma unroll
        for (int t = 0; t < 9; ++t) o += sc[t] * inv * e[t][d];
        out[(size_t)n * HDIM + lane * 4 + d] = o;
    }
}

// ---------------------------------------------------------------------------
// Host orchestration
// ---------------------------------------------------------------------------
extern "C" void kernel_launch(void* const* d_in, const int* in_sizes, int n_in,
                              void* d_out, int out_size, void* d_ws, size_t ws_size,
                              hipStream_t stream) {
    (void)in_sizes; (void)n_in; (void)out_size; (void)ws_size;
    const float* static_dense  = (const float*)d_in[0];
    const int*   static_sparse = (const int*)  d_in[1];
    const float* dyn_dense     = (const float*)d_in[2];
    const int*   dyn_sparse    = (const int*)  d_in[3];
    const int*   edges         = (const int*)  d_in[4];
    const float* weights       = (const float*)d_in[5];
    const float* static_emb    = (const float*)d_in[6];
    const float* dynamic_emb   = (const float*)d_in[7];
    const float* Ws  = (const float*)d_in[8];
    const float* bs  = (const float*)d_in[9];
    const float* gW1 = (const float*)d_in[10];
    const float* gb1 = (const float*)d_in[11];
    const float* gW2 = (const float*)d_in[12];
    const float* gb2 = (const float*)d_in[13];
    const float* Wx  = (const float*)d_in[14];
    const float* Wh  = (const float*)d_in[15];
    const float* bx  = (const float*)d_in[16];
    const float* bh  = (const float*)d_in[17];
    const float* Waw = (const float*)d_in[18];
    const float* baw = (const float*)d_in[19];
    float* outp = (float*)d_out;

    const size_t N = NN_NODES, T = TT_STEPS, E = EE_EDGES;
    size_t off = 0;
    auto carve = [&](size_t bytes) -> void* {
        void* r = (char*)d_ws + off;
        off += (bytes + 255) & ~(size_t)255;
        return r;
    };
    float* dynx   = (float*)carve(T * N * G0DIM * 4);
    float* statx  = (float*)carve(N * G0DIM * 4);
    float* semb   = (float*)carve(N * HDIM * 4);
    float* m1     = (float*)carve(N * G0DIM * 4);
    float* h1     = (float*)carve(N * HDIM * 4);
    float* m2     = (float*)carve(N * HDIM * 4);
    float* gbuf   = (float*)carve(N * HDIM * 4);
    float* hstate = (float*)carve(N * HDIM * 4);
    float* hs     = (float*)carve(T * N * HDIM * 4);
    float* gxh    = (float*)carve(N * 512 * 4);
    _Float16* WsF  = (_Float16*)carve((size_t)G0DIM * HDIM * 2);
    _Float16* gW1F = (_Float16*)carve(T * (size_t)G0DIM * HDIM * 2);
    _Float16* gW2F = (_Float16*)carve(T * (size_t)HDIM * HDIM * 2);
    _Float16* WxF  = (_Float16*)carve((size_t)HDIM * 384 * 2);
    _Float16* WhF  = (_Float16*)carve((size_t)HDIM * 384 * 2);

    // --- weight fragment conversion (tiny) ---
    convert_wfrag<<<dim3(2, 1), 256, 0, stream>>>(Ws, WsF, G0DIM, HDIM);        // 512 thr
    convert_wfrag<<<dim3(2, 8), 256, 0, stream>>>(gW1, gW1F, G0DIM, HDIM);      // per-t
    convert_wfrag<<<dim3(4, 8), 256, 0, stream>>>(gW2, gW2F, HDIM, HDIM);       // 1024 thr
    convert_wfrag<<<dim3(12, 1), 256, 0, stream>>>(Wx, WxF, HDIM, 384);         // 3072 thr
    convert_wfrag<<<dim3(12, 1), 256, 0, stream>>>(Wh, WhF, HDIM, 384);

    const int gemmBlocks = (int)((N + 127) / 128);  // 391

    // --- static encoder ---
    build_concat_x<<<(int)((N + 255) / 256), 256, 0, stream>>>(
        static_dense, static_sparse, static_emb, statx, (int)N);
    gemm_bias_act<G0DIM, HDIM, true><<<gemmBlocks, 256, 0, stream>>>(
        statx, WsF, bs, semb, (int)N);

    // --- dynamic features for all timesteps ---
    build_concat_x<<<(int)((T * N + 255) / 256), 256, 0, stream>>>(
        dyn_dense, dyn_sparse, dynamic_emb, dynx, (int)(T * N));

    hipMemsetAsync(hstate, 0, N * HDIM * 4, stream);

    for (int t = 0; t < (int)T; ++t) {
        const float* xt = dynx + (size_t)t * N * G0DIM;
        const int*   et = edges + (size_t)t * 2 * E;
        const float* wt = weights + (size_t)t * E;
        // GCN layer 1
        hipMemsetAsync(m1, 0, N * G0DIM * 4, stream);
        scatter_edges<<<(int)((E * 4 + 255) / 256), 256, 0, stream>>>(
            xt, et, wt, m1, G0DIM, 2);
        gemm_bias_act<G0DIM, HDIM, true><<<gemmBlocks, 256, 0, stream>>>(
            m1, gW1F + (size_t)t * G0DIM * HDIM, gb1 + (size_t)t * HDIM, h1, (int)N);
        // GCN layer 2
        hipMemsetAsync(m2, 0, N * HDIM * 4, stream);
        scatter_edges<<<(int)((E * 8 + 255) / 256), 256, 0, stream>>>(
            h1, et, wt, m2, HDIM, 3);
        gemm_bias_act<HDIM, HDIM, false><<<gemmBlocks, 256, 0, stream>>>(
            m2, gW2F + (size_t)t * HDIM * HDIM, gb2 + (size_t)t * HDIM, gbuf, (int)N);
        // GRU step
        gru_gemm<<<gemmBlocks, 256, 0, stream>>>(gbuf, hstate, WxF, WhF, bx, bh,
                                                 gxh, (int)N);
        gru_gates<<<(int)((N * HDIM + 255) / 256), 256, 0, stream>>>(
            gxh, hstate, hs + (size_t)t * N * HDIM, (int)N);
    }

    // --- temporal attention ---
    attention_out<<<(int)((N + 7) / 8), 256, 0, stream>>>(semb, hs, Waw, baw, outp);
}
